// BaselineModel_59708635349462
// MI455X (gfx1250) — compile-verified, matrix operand order checked by reading, not run on hardware
//
#include <hip/hip_runtime.h>

// ---------------- problem constants (match reference) ----------------
static constexpr int kNU = 200000;
static constexpr int kNI = 200000;
static constexpr int kE  = 500000;
static constexpr int kC  = 128;
static constexpr int kL  = 2;

typedef __attribute__((ext_vector_type(16))) __bf16 v16bf;
typedef __attribute__((ext_vector_type(8)))  __bf16 v8bf;
typedef __attribute__((ext_vector_type(4)))  __bf16 v4bf;
typedef __attribute__((ext_vector_type(8)))  float  v8f;

union V16U { v16bf v; v8bf h[2]; };

__device__ __forceinline__ v8f wmma_bf16(v16bf a, v16bf b, v8f c) {
    return __builtin_amdgcn_wmma_f32_16x16x32_bf16(false, a, false, b, (short)0, c, false, false);
}

// ---------------- helpers: order-preserving float<->uint for atomic max ----------------
__device__ __forceinline__ unsigned enc_f(float f) {
    int i = __float_as_int(f);
    return (i >= 0) ? ((unsigned)i | 0x80000000u) : ~(unsigned)i;
}
__device__ __forceinline__ float dec_f(unsigned u) {
    int i = (u & 0x80000000u) ? (int)(u & 0x7FFFFFFFu) : ~(int)u;
    return __int_as_float(i);
}

// ---------------- W preprocessing ----------------
// Pre-swizzle W_src into WMMA B-fragment order, split into bf16 hi/lo:
//   layout: [lr(6)][ntile(8)][ktile(4)][lane(32)][j(16)]
//   element (lane,j) of tile holds W[k][n] with
//     k = ktile*32 + (lane>=16 ? 8 : 0) + j + (j>=8 ? 8 : 0),  n = ntile*16 + (lane&15)
__global__ void swizzle_w_kernel(const float* __restrict__ W, __bf16* __restrict__ Whi,
                                 __bf16* __restrict__ Wlo) {
    int idx = blockIdx.x * blockDim.x + threadIdx.x;       // 6*8*4*32*16 = 98304 total
    if (idx >= kL * 3 * kC * kC) return;
    int j    = idx & 15;
    int lane = (idx >> 4) & 31;
    int kt   = (idx >> 9) & 3;
    int nbt  = (idx >> 11) & 7;
    int lr   = idx >> 14;
    int k = kt * 32 + ((lane >> 4) * 8) + j + ((j >= 8) ? 8 : 0);
    int n = nbt * 16 + (lane & 15);
    float f  = W[(size_t)lr * kC * kC + k * kC + n];
    __bf16 h = (__bf16)f;
    Whi[idx] = h;
    Wlo[idx] = (__bf16)(f - (float)h);
}

// vd[lr][k] = sum_c W_dst[lr][k][c] * att_dst[lr][c]   (so a_dst = x_dst . vd)
__global__ void make_vd_kernel(const float* __restrict__ Wdst, const float* __restrict__ attdst,
                               float* __restrict__ vd) {
    int lr = blockIdx.x;      // 0..5
    int k  = threadIdx.x;     // 0..127
    const float* W = Wdst + (size_t)lr * kC * kC;
    const float* a = attdst + (size_t)lr * kC;
    float s = 0.f;
    for (int c = 0; c < kC; ++c) s += W[k * kC + c] * a[c];
    vd[lr * kC + k] = s;
}

// ---------------- GEMM: H[N,128] = X[N,128] @ W[128,128], bf16-split WMMA ----------------
// 32 rows per block; 8 waves each own a 16-col N-tile and two M-tiles (2 accumulators).
__global__ __launch_bounds__(256)
void gemm_xw_kernel(const float* __restrict__ X, const __bf16* __restrict__ Wswz_hi,
                    const __bf16* __restrict__ Wswz_lo, float* __restrict__ H) {
    __shared__ alignas(16) __bf16 s_hi[32 * kC];
    __shared__ alignas(16) __bf16 s_lo[32 * kC];
    const int row0 = blockIdx.x * 32;
    const int tid  = threadIdx.x;

    // stage + split-convert 32x128 f32 tile (float4 vectorized, coalesced)
    const float4* X4 = (const float4*)(X + (size_t)row0 * kC);
    for (int i = tid; i < 32 * 32; i += 256) {
        float4 f = X4[i];
        v4bf h, l;
        h[0] = (__bf16)f.x;  h[1] = (__bf16)f.y;  h[2] = (__bf16)f.z;  h[3] = (__bf16)f.w;
        l[0] = (__bf16)(f.x - (float)h[0]);
        l[1] = (__bf16)(f.y - (float)h[1]);
        l[2] = (__bf16)(f.z - (float)h[2]);
        l[3] = (__bf16)(f.w - (float)h[3]);
        *(v4bf*)(s_hi + i * 4) = h;
        *(v4bf*)(s_lo + i * 4) = l;
    }
    __syncthreads();

    const int wave  = tid >> 5;          // N-tile
    const int lane  = tid & 31;
    const int lm    = lane & 15;
    const int khalf = (lane >> 4) * 8;

    const __bf16* wbh = Wswz_hi + ((size_t)wave * 4 * 32 + lane) * 16;
    const __bf16* wbl = Wswz_lo + ((size_t)wave * 4 * 32 + lane) * 16;
    const __bf16* a0hP = s_hi + lm * kC;
    const __bf16* a0lP = s_lo + lm * kC;
    const __bf16* a1hP = s_hi + (16 + lm) * kC;
    const __bf16* a1lP = s_lo + (16 + lm) * kC;

    v8f acc0 = {}, acc1 = {};
#pragma unroll
    for (int kt = 0; kt < 4; ++kt) {
        const int kbase = kt * 32 + khalf;
        V16U bh, bl, a0h, a0l, a1h, a1l;
        bh.v = *(const v16bf*)(wbh + (size_t)kt * 512);   // 32B aligned vector load
        bl.v = *(const v16bf*)(wbl + (size_t)kt * 512);
        a0h.h[0] = *(const v8bf*)(a0hP + kbase);
        a0h.h[1] = *(const v8bf*)(a0hP + kbase + 16);
        a0l.h[0] = *(const v8bf*)(a0lP + kbase);
        a0l.h[1] = *(const v8bf*)(a0lP + kbase + 16);
        a1h.h[0] = *(const v8bf*)(a1hP + kbase);
        a1h.h[1] = *(const v8bf*)(a1hP + kbase + 16);
        a1l.h[0] = *(const v8bf*)(a1lP + kbase);
        a1l.h[1] = *(const v8bf*)(a1lP + kbase + 16);

        acc0 = wmma_bf16(a0h.v, bh.v, acc0);
        acc0 = wmma_bf16(a0h.v, bl.v, acc0);
        acc0 = wmma_bf16(a0l.v, bh.v, acc0);
        acc1 = wmma_bf16(a1h.v, bh.v, acc1);
        acc1 = wmma_bf16(a1h.v, bl.v, acc1);
        acc1 = wmma_bf16(a1l.v, bh.v, acc1);
    }

    // D layout: VGPR v -> M = v + (lane>=16 ? 8 : 0), N = nb + (lane&15). No bounds checks (N % 32 == 0).
    const int moff = (lane >> 4) * 8;
    float* H0 = H + (size_t)(row0 + moff) * kC + wave * 16 + lm;
    float* H1 = H + (size_t)(row0 + 16 + moff) * kC + wave * 16 + lm;
#pragma unroll
    for (int v = 0; v < 8; ++v) H0[(size_t)v * kC] = acc0[v];
#pragma unroll
    for (int v = 0; v < 8; ++v) H1[(size_t)v * kC] = acc1[v];
}

// ---------------- row dot: out[n] = M[n,:] . v  (one warp per row, float4 lanes) ----------------
__global__ __launch_bounds__(256)
void rowdot_kernel(const float* __restrict__ M, const float* __restrict__ v,
                   float* __restrict__ out, int n) {
    int row  = blockIdx.x * 8 + (threadIdx.x >> 5);
    int lane = threadIdx.x & 31;
    if (row >= n) return;
    float4 m4 = *(const float4*)(M + (size_t)row * kC + lane * 4);
    float4 v4 = *(const float4*)(v + lane * 4);
    float s = m4.x * v4.x + m4.y * v4.y + m4.z * v4.z + m4.w * v4.w;
    for (int off = 16; off > 0; off >>= 1) s += __shfl_down(s, off, 32);
    if (lane == 0) out[row] = s;
}

// ---------------- per-relation init ----------------
__global__ void init_seg_kernel(unsigned* __restrict__ mbuf, float* __restrict__ denom, int n) {
    int i = blockIdx.x * blockDim.x + threadIdx.x;
    if (i >= n) return;
    mbuf[i]  = enc_f(-__builtin_huge_valf());
    denom[i] = 0.f;
}
__global__ void init_out_bias_kernel(float* __restrict__ out, const float* __restrict__ bias, int n) {
    int i = blockIdx.x * blockDim.x + threadIdx.x;
    if (i >= n * kC) return;
    out[i] = bias[i & 127];
}
__global__ void add_bias_kernel(float* __restrict__ out, const float* __restrict__ bias, int n) {
    int i = blockIdx.x * blockDim.x + threadIdx.x;
    if (i >= n * kC) return;
    out[i] += bias[i & 127];
}

// ---------------- edge passes ----------------
__global__ void edge_logits_kernel(const int* __restrict__ ei, const float* __restrict__ asrc,
                                   const float* __restrict__ adst, float* __restrict__ ebuf,
                                   unsigned* __restrict__ mbuf, int ne) {
    int e = blockIdx.x * blockDim.x + threadIdx.x;
    if (e >= ne) return;
    int src = ei[e], dst = ei[ne + e];
    float x = asrc[src] + adst[dst];
    x = (x > 0.f) ? x : 0.2f * x;   // leaky relu, slope 0.2
    ebuf[e] = x;
    atomicMax(&mbuf[dst], enc_f(x));
}

__global__ void edge_exp_kernel(const int* __restrict__ ei, float* __restrict__ ebuf,
                                const unsigned* __restrict__ mbuf, float* __restrict__ denom, int ne) {
    int e = blockIdx.x * blockDim.x + threadIdx.x;
    if (e >= ne) return;
    int dst = ei[ne + e];
    float w = __expf(ebuf[e] - dec_f(mbuf[dst]));
    ebuf[e] = w;
    atomicAdd(&denom[dst], w);
}

// one warp per edge: out[dst,:] += hs[src,:] * (w / denom[dst])
__global__ __launch_bounds__(256)
void edge_scatter_kernel(const int* __restrict__ ei, const float* __restrict__ hs,
                         const float* __restrict__ wbuf, const float* __restrict__ denom,
                         float* __restrict__ out, int ne) {
    int e    = blockIdx.x * 8 + (threadIdx.x >> 5);
    int lane = threadIdx.x & 31;
    if (e >= ne) return;
    int src = ei[e], dst = ei[ne + e];
    float alpha = wbuf[e] / denom[dst];
    float4 h4 = *(const float4*)(hs + (size_t)src * kC + lane * 4);
    float* o = out + (size_t)dst * kC + lane * 4;
    atomicAdd(o + 0, h4.x * alpha);
    atomicAdd(o + 1, h4.y * alpha);
    atomicAdd(o + 2, h4.z * alpha);
    atomicAdd(o + 3, h4.w * alpha);
}

// ---------------- LayerNorm + ReLU, one warp per node (float4 lanes) ----------------
__global__ __launch_bounds__(256)
void ln_relu_kernel(const float* __restrict__ in, const float* __restrict__ w,
                    const float* __restrict__ b, float* __restrict__ out, int n) {
    int row  = blockIdx.x * 8 + (threadIdx.x >> 5);
    int lane = threadIdx.x & 31;
    if (row >= n) return;
    float4 v  = *(const float4*)(in + (size_t)row * kC + lane * 4);
    float s = v.x + v.y + v.z + v.w;
    for (int off = 16; off > 0; off >>= 1) s += __shfl_xor(s, off, 32);
    float mu = s * (1.0f / kC);
    float dx = v.x - mu, dy = v.y - mu, dz = v.z - mu, dw = v.w - mu;
    float var = dx * dx + dy * dy + dz * dz + dw * dw;
    for (int off = 16; off > 0; off >>= 1) var += __shfl_xor(var, off, 32);
    var *= (1.0f / kC);
    float inv = rsqrtf(var + 1e-5f);
    float4 w4 = *(const float4*)(w + lane * 4);
    float4 b4 = *(const float4*)(b + lane * 4);
    float4 y;
    y.x = fmaxf(dx * inv * w4.x + b4.x, 0.f);
    y.y = fmaxf(dy * inv * w4.y + b4.y, 0.f);
    y.z = fmaxf(dz * inv * w4.z + b4.z, 0.f);
    y.w = fmaxf(dw * inv * w4.w + b4.w, 0.f);
    *(float4*)(out + (size_t)row * kC + lane * 4) = y;
}

// ---------------- host orchestration ----------------
static inline size_t align256(size_t x) { return (x + 255) & ~size_t(255); }

extern "C" void kernel_launch(void* const* d_in, const int* in_sizes, int n_in,
                              void* d_out, int out_size, void* d_ws, size_t ws_size,
                              hipStream_t stream) {
    (void)in_sizes; (void)n_in; (void)out_size; (void)ws_size;

    const float* x_user  = (const float*)d_in[0];
    const float* x_item  = (const float*)d_in[1];
    const int*   ei_u2i  = (const int*)d_in[2];
    const int*   ei_i2u  = (const int*)d_in[3];
    const int*   ei_u2u  = (const int*)d_in[4];
    const float* W_src   = (const float*)d_in[5];   // [L,3,C,C]
    const float* W_dst   = (const float*)d_in[6];   // [L,3,C,C]
    const float* att_src = (const float*)d_in[7];   // [L,3,C]
    const float* att_dst = (const float*)d_in[8];   // [L,3,C]
    const float* gbias   = (const float*)d_in[9];   // [L,3,C]
    const float* ln_w    = (const float*)d_in[10];  // [L,2,C]
    const float* ln_b    = (const float*)d_in[11];  // [L,2,C]
    float* outp = (float*)d_out;                    // [(NU+NI), C]

    // ---- carve workspace ----
    char* p = (char*)d_ws;
    const size_t featBytes = (size_t)kNU * kC * sizeof(float);   // NU == NI
    float*    hs     = (float*)p;     p += align256(featBytes);
    float*    out_u  = (float*)p;     p += align256(featBytes);
    float*    out_i  = (float*)p;     p += align256(featBytes);
    float*    xu_buf = (float*)p;     p += align256(featBytes);
    float*    xi_buf = (float*)p;     p += align256(featBytes);
    float*    asrc   = (float*)p;     p += align256((size_t)kNU * sizeof(float));
    float*    adst   = (float*)p;     p += align256((size_t)kNU * sizeof(float));
    unsigned* mbuf   = (unsigned*)p;  p += align256((size_t)kNU * sizeof(unsigned));
    float*    denom  = (float*)p;     p += align256((size_t)kNU * sizeof(float));
    float*    ebuf   = (float*)p;     p += align256((size_t)kE  * sizeof(float));
    __bf16*   Whi    = (__bf16*)p;    p += align256((size_t)kL * 3 * kC * kC * sizeof(__bf16));
    __bf16*   Wlo    = (__bf16*)p;    p += align256((size_t)kL * 3 * kC * kC * sizeof(__bf16));
    float*    vd     = (float*)p;     p += align256((size_t)kL * 3 * kC * sizeof(float));

    // ---- per-call preprocessing (cheap, deterministic) ----
    {
        int nW = kL * 3 * kC * kC;  // 98304
        swizzle_w_kernel<<<(nW + 255) / 256, 256, 0, stream>>>(W_src, Whi, Wlo);
        make_vd_kernel<<<kL * 3, kC, 0, stream>>>(W_dst, att_dst, vd);
    }

    const float* xu = x_user;
    const float* xi = x_item;

    for (int l = 0; l < kL; ++l) {
        const float* srcX[3] = { xu, xi, xu };
        const float* dstX[3] = { xi, xu, xu };
        const int*   eis[3]  = { ei_u2i, ei_i2u, ei_u2u };
        const int    nsrc[3] = { kNU, kNI, kNU };
        const int    ndst[3] = { kNI, kNU, kNU };
        float*       outb[3] = { out_i, out_u, out_u };

        for (int r = 0; r < 3; ++r) {
            const int lr = l * 3 + r;
            const int ns = nsrc[r], nd = ndst[r];

            // hs = x_src @ W_src[l,r]  (WMMA bf16-split; 32 rows/block)
            gemm_xw_kernel<<<ns / 32, 256, 0, stream>>>(
                srcX[r], Whi + (size_t)lr * kC * kC, Wlo + (size_t)lr * kC * kC, hs);

            // a_src[n] = hs[n,:] . att_src[l,r] ; a_dst[n] = x_dst[n,:] . (W_dst@att_dst)
            rowdot_kernel<<<(ns + 7) / 8, 256, 0, stream>>>(hs, att_src + (size_t)lr * kC, asrc, ns);
            rowdot_kernel<<<(nd + 7) / 8, 256, 0, stream>>>(dstX[r], vd + (size_t)lr * kC, adst, nd);

            // segment state + output init (rel2 accumulates onto rel1's output)
            init_seg_kernel<<<(nd + 255) / 256, 256, 0, stream>>>(mbuf, denom, nd);
            if (r == 2)
                add_bias_kernel<<<((size_t)nd * kC + 255) / 256, 256, 0, stream>>>(outb[r], gbias + (size_t)lr * kC, nd);
            else
                init_out_bias_kernel<<<((size_t)nd * kC + 255) / 256, 256, 0, stream>>>(outb[r], gbias + (size_t)lr * kC, nd);

            // edge softmax + weighted scatter
            edge_logits_kernel<<<(kE + 255) / 256, 256, 0, stream>>>(eis[r], asrc, adst, ebuf, mbuf, kE);
            edge_exp_kernel<<<(kE + 255) / 256, 256, 0, stream>>>(eis[r], ebuf, mbuf, denom, kE);
            edge_scatter_kernel<<<(kE + 7) / 8, 256, 0, stream>>>(eis[r], hs, ebuf, denom, outb[r], kE);
        }

        // LayerNorm + ReLU -> next features (last layer writes straight into d_out)
        float* xu_next = (l == kL - 1) ? outp : xu_buf;
        float* xi_next = (l == kL - 1) ? (outp + (size_t)kNU * kC) : xi_buf;
        ln_relu_kernel<<<(kNU + 7) / 8, 256, 0, stream>>>(out_u, ln_w + (size_t)(l * 2 + 0) * kC,
                                                          ln_b + (size_t)(l * 2 + 0) * kC, xu_next, kNU);
        ln_relu_kernel<<<(kNI + 7) / 8, 256, 0, stream>>>(out_i, ln_w + (size_t)(l * 2 + 1) * kC,
                                                          ln_b + (size_t)(l * 2 + 1) * kC, xi_next, kNI);
        xu = xu_buf;
        xi = xi_buf;
    }
}
